// SwinTransformerBlock_63419487093246
// MI455X (gfx1250) — matrix operand order repaired and checked by reading, not compiled, not run on hardware
//
#include <hip/hip_runtime.h>
#include <math.h>

typedef __bf16 bf16;
typedef __attribute__((ext_vector_type(16))) __bf16 v16bf;
typedef __attribute__((ext_vector_type(8)))  float  v8f;
typedef unsigned int v4u __attribute__((ext_vector_type(4)));
typedef int          v8i __attribute__((ext_vector_type(8)));
typedef int          v4i __attribute__((ext_vector_type(4)));

#define DIM 192
#define HEADS 6
#define HD 32
#define NTOK 64          // tokens per window
#define NWIN 1024        // windows per image (32x32)
#define BATCH 4
#define MTOT 262144      // BATCH*H*W tokens
#define QKVC 576
#define MLPH 768
#define SCALE_ 0.17677669529663687f

#if defined(__HIP_DEVICE_COMPILE__) && __has_builtin(__builtin_amdgcn_tensor_load_to_lds)
#define HAVE_TDM 1
#else
#define HAVE_TDM 0
#endif

union FragU { v16bf v; uint4 q[2]; };

__device__ __forceinline__ v16bf frag_a(const bf16* p) {
    // 16-bit A 16x32 layout: lane half 0 -> K 0..7 & 16..23 ; half 1 -> K 8..15 & 24..31
    FragU f;
    f.q[0] = *(const uint4*)p;
    f.q[1] = *(const uint4*)(p + 16);
    return f.v;
}
__device__ __forceinline__ v16bf frag_b(const bf16* p) {
    // 16-bit B 32x16 layout: lane half 0 -> K 0..7 & 16..23 ; half 1 -> K 8..15 & 24..31
    FragU f;
    f.q[0] = *(const uint4*)p;
    f.q[1] = *(const uint4*)(p + 16);
    return f.v;
}

#if HAVE_TDM
// Issue one 2-D TDM tile load: tile_d1 x tile_d0 (rows x K) of bf16 from a
// row-major tensor with row stride `stride_e` elements, packed contiguously
// into LDS at byte offset lds_off.  D# per CDNA5 ISA 8.3/8.4.
// This toolchain's builtin takes 6 args: (g0 v4u, g1 v8i, g2 v4i, g3 v4i, gx v8i, cpol).
__device__ __forceinline__ void tdm_load_2d(const bf16* gsrc, unsigned lds_off,
                                            unsigned ten_d0, unsigned ten_d1,
                                            unsigned til_d0, unsigned til_d1,
                                            unsigned stride_e) {
    unsigned long long ga = (unsigned long long)(uintptr_t)gsrc;
    v4u g0;
    g0[0] = 1u;                                   // count=1, user descriptor
    g0[1] = lds_off;                              // lds_addr (bytes)
    g0[2] = (unsigned)(ga & 0xffffffffu);         // global_addr[31:0]
    g0[3] = (unsigned)((ga >> 32) & 0x1ffffffu)   // global_addr[56:32]
            | 0x80000000u;                        // type=2 ("image")
    v8i g1;
    g1[0] = (int)(1u << 16);                      // wg_mask=0, data_size=1 (2B)
    g1[1] = (int)((ten_d0 & 0xffffu) << 16);      // [63:48]=tensor_dim0 lo
    g1[2] = (int)(((ten_d0 >> 16) & 0xffffu) | ((ten_d1 & 0xffffu) << 16));
    g1[3] = (int)(((ten_d1 >> 16) & 0xffffu) | ((til_d0 & 0xffffu) << 16));
    g1[4] = (int)(til_d1 & 0xffffu);              // tile_dim1; tile_dim2=0
    g1[5] = (int)stride_e;                        // tensor_dim0_stride[31:0]
    g1[6] = 0;                                    // stride hi / dim1_stride
    g1[7] = 0;
    v4i z4 = {0, 0, 0, 0};                        // groups 2/3 unused (2-D)
    v8i z8 = {0, 0, 0, 0, 0, 0, 0, 0};
    __builtin_amdgcn_tensor_load_to_lds(g0, g1, z4, z4, z8, 0);
}
#endif

// ---------------------------------------------------------------- fp32->bf16
__global__ __launch_bounds__(256) void f2b_kernel(const float* __restrict__ a,
                                                  bf16* __restrict__ o, int n) {
    int i = blockIdx.x * 256 + threadIdx.x;
    if (i < n) o[i] = (bf16)a[i];
}
// ------------------------------------------- fp32 (KxN) -> bf16 transposed (NxK)
__global__ __launch_bounds__(256) void wtr_kernel(const float* __restrict__ w,
                                                  bf16* __restrict__ wt, int K, int N) {
    int i = blockIdx.x * 256 + threadIdx.x;
    if (i < K * N) {
        int n = i / K, k = i - n * K;
        wt[i] = (bf16)w[(size_t)k * N + n];
    }
}

// ------------------------------------------- LN1 + cyclic shift + window part
__global__ __launch_bounds__(64)
void ln1_window_kernel(const float* __restrict__ x, const float* __restrict__ gw,
                       const float* __restrict__ gb, bf16* __restrict__ out) {
    const int t = threadIdx.x;
    const int bid = blockIdx.x;            // destination window-token id
    const int n = bid & 63, win = bid >> 6;
    const int wi = win & (NWIN - 1), bb = win >> 10;
    const int hs = ((wi >> 5) << 3) + (n >> 3);
    const int ws = ((wi & 31) << 3) + (n & 7);
    const int h = (hs + 4) & 255, w = (ws + 4) & 255;   // roll(-SHIFT)
    const float* row = x + ((size_t)bb * 65536 + (size_t)h * 256 + w) * DIM;
    float v0 = row[t], v1 = row[t + 64], v2 = row[t + 128];
    float s = v0 + v1 + v2;
#pragma unroll
    for (int m = 16; m >= 1; m >>= 1) s += __shfl_xor(s, m, 32);
    __shared__ float sm[2], sq[2];
    if ((t & 31) == 0) sm[t >> 5] = s;
    __syncthreads();
    float mu = (sm[0] + sm[1]) * (1.0f / DIM);
    float d0 = v0 - mu, d1 = v1 - mu, d2 = v2 - mu;
    float q = d0 * d0 + d1 * d1 + d2 * d2;
#pragma unroll
    for (int m = 16; m >= 1; m >>= 1) q += __shfl_xor(q, m, 32);
    if ((t & 31) == 0) sq[t >> 5] = q;
    __syncthreads();
    float inv = rsqrtf((sq[0] + sq[1]) * (1.0f / DIM) + 1e-5f);
    bf16* orow = out + (size_t)bid * DIM;
    orow[t]       = (bf16)(d0 * inv * gw[t]       + gb[t]);
    orow[t + 64]  = (bf16)(d1 * inv * gw[t + 64]  + gb[t + 64]);
    orow[t + 128] = (bf16)(d2 * inv * gw[t + 128] + gb[t + 128]);
}

// --------------------------------------------------------------- plain LN2
__global__ __launch_bounds__(64)
void ln2_kernel(const float* __restrict__ x, const float* __restrict__ gw,
                const float* __restrict__ gb, bf16* __restrict__ out) {
    const int t = threadIdx.x;
    const float* row = x + (size_t)blockIdx.x * DIM;
    float v0 = row[t], v1 = row[t + 64], v2 = row[t + 128];
    float s = v0 + v1 + v2;
#pragma unroll
    for (int m = 16; m >= 1; m >>= 1) s += __shfl_xor(s, m, 32);
    __shared__ float sm[2], sq[2];
    if ((t & 31) == 0) sm[t >> 5] = s;
    __syncthreads();
    float mu = (sm[0] + sm[1]) * (1.0f / DIM);
    float d0 = v0 - mu, d1 = v1 - mu, d2 = v2 - mu;
    float q = d0 * d0 + d1 * d1 + d2 * d2;
#pragma unroll
    for (int m = 16; m >= 1; m >>= 1) q += __shfl_xor(q, m, 32);
    if ((t & 31) == 0) sq[t >> 5] = q;
    __syncthreads();
    float inv = rsqrtf((sq[0] + sq[1]) * (1.0f / DIM) + 1e-5f);
    bf16* orow = out + (size_t)blockIdx.x * DIM;
    orow[t]       = (bf16)(d0 * inv * gw[t]       + gb[t]);
    orow[t + 64]  = (bf16)(d1 * inv * gw[t + 64]  + gb[t + 64]);
    orow[t + 128] = (bf16)(d2 * inv * gw[t + 128] + gb[t + 128]);
}

// ---------------------------------------------------------------- WMMA GEMM
// C = A(bf16, MxK) * Wt^T + bias, with Wt given TRANSPOSED (N x K, bf16).
// OUT_MODE: 0=bf16 store, 2=fp32 add-to-C, 3=fp32 window-reverse + residual.
#define BM 128
#define BN 64
#define BK 32

template <int OUT_MODE, bool GELU>
__global__ __launch_bounds__(256)
void gemm_bf16_wmma(const bf16* __restrict__ A, const bf16* __restrict__ Wt,
                    const float* __restrict__ bias, void* __restrict__ C,
                    const float* __restrict__ resid, int M, int K, int N) {
    __shared__ __align__(16) bf16 As[2][BM * BK];   // [m][k], double buffered
    __shared__ __align__(16) bf16 Bs[2][BN * BK];   // [n][k] (Wt rows)
    const int tid = threadIdx.x;
    const int lane = tid & 31;
    const int wid = tid >> 5;
    const int nbN = N / BN;
    const int bm = (blockIdx.x / nbN) * BM;
    const int bn = (blockIdx.x % nbN) * BN;
    const int wm = (wid >> 1) * 32;
    const int wn = (wid & 1) * 32;
    const int lhalf = lane >> 4, l15 = lane & 15;
    const int nk = K / BK;

    v8f acc[2][2] = {};

#if HAVE_TDM
    if (wid == 0) {   // TDM DMA prologue: stage K-tile 0
        tdm_load_2d(A + (size_t)bm * K, (unsigned)(uintptr_t)&As[0][0],
                    K, M - bm, BK, BM, K);
        tdm_load_2d(Wt + (size_t)bn * K, (unsigned)(uintptr_t)&Bs[0][0],
                    K, N - bn, BK, BN, K);
    }
    for (int it = 0; it < nk; ++it) {
        const int kt = it * BK;
        const int cur = it & 1;
        if (wid == 0) __builtin_amdgcn_s_wait_tensorcnt(0);
        __syncthreads();                       // tile `cur` ready, prev reads done
        if (wid == 0 && it + 1 < nk) {         // overlap DMA of next tile
            tdm_load_2d(A + (size_t)bm * K + kt + BK,
                        (unsigned)(uintptr_t)&As[cur ^ 1][0],
                        K - kt - BK, M - bm, BK, BM, K);
            tdm_load_2d(Wt + (size_t)bn * K + kt + BK,
                        (unsigned)(uintptr_t)&Bs[cur ^ 1][0],
                        K - kt - BK, N - bn, BK, BN, K);
        }
#else
    for (int it = 0; it < nk; ++it) {
        const int kt = it * BK;
        const int cur = 0;
        {   // A tile: 512 x 16B chunks, 2 per thread
#pragma unroll
            for (int c0 = 0; c0 < 2; ++c0) {
                int c = tid + c0 * 256;
                int row = c >> 2, kc = (c & 3) * 8;
                *(uint4*)(&As[0][row * BK + kc]) =
                    *(const uint4*)(A + (size_t)(bm + row) * K + kt + kc);
            }
            // B tile: one 16B chunk per thread from transposed weights
            int n = tid >> 2, k0 = (tid & 3) * 8;
            *(uint4*)(&Bs[0][n * BK + k0]) =
                *(const uint4*)(Wt + (size_t)(bn + n) * K + kt + k0);
        }
        if (kt + BK < K)
            __builtin_prefetch(A + (size_t)(bm + (tid >> 1)) * K + kt + BK, 0, 1);
        __syncthreads();
#endif
        v16bf af[2], bfv[2];
#pragma unroll
        for (int i = 0; i < 2; ++i)
            af[i] = frag_a(&As[cur][(wm + i * 16 + l15) * BK + lhalf * 8]);
#pragma unroll
        for (int j = 0; j < 2; ++j)
            bfv[j] = frag_b(&Bs[cur][(wn + j * 16 + l15) * BK + lhalf * 8]);
#pragma unroll
        for (int i = 0; i < 2; ++i)
#pragma unroll
            for (int j = 0; j < 2; ++j)
                acc[i][j] = __builtin_amdgcn_wmma_f32_16x16x32_bf16(
                    false, af[i], false, bfv[j], (short)0, acc[i][j], false, false);
#if !HAVE_TDM
        __syncthreads();
#endif
    }

#pragma unroll
    for (int i = 0; i < 2; ++i)
#pragma unroll
        for (int j = 0; j < 2; ++j) {
            int col = bn + wn + j * 16 + l15;
            float bv = bias[col];
#pragma unroll
            for (int r = 0; r < 8; ++r) {
                int row = bm + wm + i * 16 + r + 8 * lhalf;
                float v = acc[i][j][r] + bv;
                if (GELU) v = 0.5f * v * (1.0f + erff(v * 0.70710678118654752f));
                if (OUT_MODE == 0) {
                    ((bf16*)C)[(size_t)row * N + col] = (bf16)v;
                } else if (OUT_MODE == 2) {
                    float* Cp = (float*)C;
                    size_t idx = (size_t)row * N + col;
                    Cp[idx] = Cp[idx] + v;
                } else {  // window reverse + roll(+SHIFT) + residual add
                    int winr = row >> 6, ntok = row & 63;
                    int wi2 = winr & (NWIN - 1), bb = winr >> 10;
                    int hs = ((wi2 >> 5) << 3) + (ntok >> 3);
                    int wss = ((wi2 & 31) << 3) + (ntok & 7);
                    int hh = (hs + 4) & 255, ww2 = (wss + 4) & 255;
                    size_t dest =
                        ((size_t)bb * 65536 + (size_t)hh * 256 + ww2) * DIM + col;
                    ((float*)C)[dest] = resid[dest] + v;
                }
            }
        }
}

// ------------------------------------------------------- windowed attention
__global__ __launch_bounds__(32)
void attn_kernel(const bf16* __restrict__ qkv, const float* __restrict__ rpb,
                 bf16* __restrict__ o) {
    const int head = blockIdx.x % HEADS;
    const int win = blockIdx.x / HEADS;
    const int wi = win & (NWIN - 1);
    const int lane = threadIdx.x & 31;
    const int lhalf = lane >> 4, l15 = lane & 15;

    __shared__ __align__(16) bf16 v_lds[HD * NTOK];    // [d][m]
    __shared__ __align__(16) bf16 p_lds[NTOK * NTOK];  // probs [m][n]
    __shared__ float rpb_lds[225 * HEADS];             // relative-pos bias table

    // stage the 5.4KB rpb table into LDS (kills scattered global gathers)
    for (int i = lane; i < 225 * HEADS; i += 32) rpb_lds[i] = rpb[i];

    const bf16* qg = qkv + (size_t)win * NTOK * QKVC + head * HD;
    const bf16* kg = qg + DIM;
    const bf16* vg = qg + 2 * DIM;

    // stage V transposed: each lane owns 2 token rows
#pragma unroll
    for (int rr = 0; rr < 2; ++rr) {
        int m = rr * 32 + lane;
        union { uint4 q[4]; bf16 h[32]; } u;
#pragma unroll
        for (int c = 0; c < 4; ++c)
            u.q[c] = *(const uint4*)(vg + (size_t)m * QKVC + c * 8);
#pragma unroll
        for (int d = 0; d < HD; ++d) v_lds[d * NTOK + m] = u.h[d];
    }

    // S = Q * K^T  (64x64, K-dim = 32 -> one WMMA per 16x16 tile)
    v16bf aq[4], bk[4];
#pragma unroll
    for (int i = 0; i < 4; ++i)
        aq[i] = frag_a(qg + (size_t)(i * 16 + l15) * QKVC + lhalf * 8);
#pragma unroll
    for (int j = 0; j < 4; ++j)
        bk[j] = frag_b(kg + (size_t)(j * 16 + l15) * QKVC + lhalf * 8);
    v8f s[4][4];
#pragma unroll
    for (int i = 0; i < 4; ++i)
#pragma unroll
        for (int j = 0; j < 4; ++j) {
            v8f z = {};
            s[i][j] = __builtin_amdgcn_wmma_f32_16x16x32_bf16(
                false, aq[i], false, bk[j], (short)0, z, false, false);
        }

    // per-lane column attributes (invariant over i,r)
    const int wh = wi >> 5, ww = wi & 31;
    int yn_[4], xn_[4], rgn_[4];
#pragma unroll
    for (int j = 0; j < 4; ++j) {
        int n = j * 16 + l15;
        yn_[j] = n >> 3; xn_[j] = n & 7;
        int hn = wh * 8 + yn_[j], wn_ = ww * 8 + xn_[j];
        rgn_[j] = (hn < 248 ? 0 : (hn < 252 ? 1 : 2)) * 3 +
                  (wn_ < 248 ? 0 : (wn_ < 252 ? 1 : 2));
    }

    // bias + shifted-window mask + softmax (rows span 16-lane halves)
#pragma unroll
    for (int i = 0; i < 4; ++i) {
#pragma unroll
        for (int r = 0; r < 8; ++r) {
            int m = i * 16 + r + 8 * lhalf;
            int ym = m >> 3, xm = m & 7;
            int hm = wh * 8 + ym, wm_ = ww * 8 + xm;
            int rgm = (hm < 248 ? 0 : (hm < 252 ? 1 : 2)) * 3 +
                      (wm_ < 248 ? 0 : (wm_ < 252 ? 1 : 2));
            float val[4];
#pragma unroll
            for (int j = 0; j < 4; ++j) {
                int ridx = (ym - yn_[j] + 7) * 15 + (xm - xn_[j] + 7);
                float bias = rpb_lds[ridx * HEADS + head];
                float mask = (rgm == rgn_[j]) ? 0.0f : -100.0f;
                val[j] = s[i][j][r] * SCALE_ + bias + mask;
            }
            float mx = fmaxf(fmaxf(val[0], val[1]), fmaxf(val[2], val[3]));
#pragma unroll
            for (int t = 1; t < 16; t <<= 1) mx = fmaxf(mx, __shfl_xor(mx, t, 32));
            float sum = 0.0f;
#pragma unroll
            for (int j = 0; j < 4; ++j) { val[j] = __expf(val[j] - mx); sum += val[j]; }
#pragma unroll
            for (int t = 1; t < 16; t <<= 1) sum += __shfl_xor(sum, t, 32);
            float inv = 1.0f / sum;
#pragma unroll
            for (int j = 0; j < 4; ++j)
                p_lds[m * NTOK + j * 16 + l15] = (bf16)(val[j] * inv);
        }
    }
    __syncthreads();

    // O = P * V  (64x32, K-dim = 64 -> two WMMA steps per tile)
    v8f oacc[4][2] = {};
#pragma unroll
    for (int kk = 0; kk < 2; ++kk) {
        v16bf ap[4], bv[2];
#pragma unroll
        for (int i = 0; i < 4; ++i)
            ap[i] = frag_a(&p_lds[(i * 16 + l15) * NTOK + kk * 32 + lhalf * 8]);
#pragma unroll
        for (int j = 0; j < 2; ++j)
            bv[j] = frag_b(&v_lds[(j * 16 + l15) * NTOK + kk * 32 + lhalf * 8]);
#pragma unroll
        for (int i = 0; i < 4; ++i)
#pragma unroll
            for (int j = 0; j < 2; ++j)
                oacc[i][j] = __builtin_amdgcn_wmma_f32_16x16x32_bf16(
                    false, ap[i], false, bv[j], (short)0, oacc[i][j], false, false);
    }
#pragma unroll
    for (int i = 0; i < 4; ++i)
#pragma unroll
        for (int j = 0; j < 2; ++j)
#pragma unroll
            for (int r = 0; r < 8; ++r) {
                int m = i * 16 + r + 8 * lhalf;
                int d = j * 16 + l15;
                o[((size_t)win * NTOK + m) * DIM + head * HD + d] = (bf16)oacc[i][j][r];
            }
}

// ------------------------------------------------------------------- driver
extern "C" void kernel_launch(void* const* d_in, const int* in_sizes, int n_in,
                              void* d_out, int out_size, void* d_ws, size_t ws_size,
                              hipStream_t stream) {
    (void)in_sizes; (void)n_in; (void)out_size; (void)ws_size;
    const float* x      = (const float*)d_in[0];
    const float* n1w    = (const float*)d_in[1];
    const float* n1b    = (const float*)d_in[2];
    const float* qkv_w  = (const float*)d_in[3];
    const float* qkv_b  = (const float*)d_in[4];
    const float* rpb    = (const float*)d_in[5];
    const float* proj_w = (const float*)d_in[6];
    const float* proj_b = (const float*)d_in[7];
    const float* n2w    = (const float*)d_in[8];
    const float* n2b    = (const float*)d_in[9];
    const float* fc1_w  = (const float*)d_in[10];
    const float* fc1_b  = (const float*)d_in[11];
    const float* fc2_w  = (const float*)d_in[12];
    const float* fc2_b  = (const float*)d_in[13];
    float* out = (float*)d_out;
    char* ws = (char*)d_ws;

    // workspace layout (bytes) -- all weights stored TRANSPOSED (N x K) bf16
    bf16* wqkv  = (bf16*)(ws);                      // 576x192
    bf16* wproj = (bf16*)(ws + 221184);             // 192x192
    bf16* wfc1  = (bf16*)(ws + 294912);             // 768x192
    bf16* wfc2  = (bf16*)(ws + 589824);             // 192x768
    const size_t OFF0 = 1ull << 20;
    bf16* xw     = (bf16*)(ws + OFF0);                               // 96 MB (xw, later x2)
    bf16* qkvbuf = (bf16*)(ws + OFF0 + 100663296ull);                // 384 MB (qkv, later mlp hidden)
    bf16* attn_o = (bf16*)(ws + OFF0 + 100663296ull + 402653184ull); // 96 MB

    wtr_kernel<<<(192 * 576 + 255) / 256, 256, 0, stream>>>(qkv_w, wqkv, 192, 576);
    wtr_kernel<<<(192 * 192 + 255) / 256, 256, 0, stream>>>(proj_w, wproj, 192, 192);
    wtr_kernel<<<(192 * 768 + 255) / 256, 256, 0, stream>>>(fc1_w, wfc1, 192, 768);
    wtr_kernel<<<(768 * 192 + 255) / 256, 256, 0, stream>>>(fc2_w, wfc2, 768, 192);

    // LN1 + shift + window partition -> bf16
    ln1_window_kernel<<<MTOT, 64, 0, stream>>>(x, n1w, n1b, xw);

    // qkv = xw @ Wqkv + b
    gemm_bf16_wmma<0, false><<<(MTOT / BM) * (QKVC / BN), 256, 0, stream>>>(
        xw, wqkv, qkv_b, qkvbuf, nullptr, MTOT, DIM, QKVC);

    // per-(window, head) attention
    attn_kernel<<<BATCH * NWIN * HEADS, 32, 0, stream>>>(qkvbuf, rpb, attn_o);

    // proj + window-reverse + roll + residual -> d_out (fp32 x1)
    gemm_bf16_wmma<3, false><<<(MTOT / BM) * (DIM / BN), 256, 0, stream>>>(
        attn_o, wproj, proj_b, out, x, MTOT, DIM, DIM);

    // LN2 -> bf16 (reuse xw region)
    ln2_kernel<<<MTOT, 64, 0, stream>>>(out, n2w, n2b, xw);

    // fc1 + GELU -> bf16 (reuse qkv region)
    gemm_bf16_wmma<0, true><<<(MTOT / BM) * (MLPH / BN), 256, 0, stream>>>(
        xw, wfc1, fc1_b, qkvbuf, nullptr, MTOT, DIM, MLPH);

    // fc2 + residual add into d_out
    gemm_bf16_wmma<2, false><<<(MTOT / BM) * (DIM / BN), 256, 0, stream>>>(
        qkvbuf, wfc2, fc2_b, out, nullptr, MTOT, MLPH, DIM);
}
